// Attention_2602750181433
// MI455X (gfx1250) — matrix operand order repaired and checked by reading, not compiled, not run on hardware
//
#include <hip/hip_runtime.h>

// Problem dims (fixed by reference)
#define B_ 4
#define M_ 256
#define N_ 512
#define D_ 512
#define H_ 512

typedef __attribute__((ext_vector_type(16))) __bf16 v16bf;
typedef __attribute__((ext_vector_type(8)))  __bf16 v8bf;
typedef __attribute__((ext_vector_type(8)))  float  v8f;

// fp32 -> bf16 bits (round-to-nearest-even)
__device__ __forceinline__ unsigned short f2bfu(float f) {
    union { float f; unsigned u; } c; c.f = f;
    unsigned r = c.u + 0x7FFFu + ((c.u >> 16) & 1u);
    return (unsigned short)(r >> 16);
}

// fast tanh via v_exp_f32 / v_rcp_f32: tanh(x) = 1 - 2/(exp2(2*log2e*x)+1)
__device__ __forceinline__ float tanh_fast(float x) {
    float e = __builtin_amdgcn_exp2f(x * 2.885390082f);
    return 1.f - 2.f * __builtin_amdgcn_rcpf(e + 1.f);
}

// CDNA5 async global -> LDS copy (ASYNCcnt-tracked), 16B per lane.
// On AMDGCN the low 32 bits of a generic LDS pointer are the LDS byte offset.
__device__ __forceinline__ void async_copy_b128(unsigned lds_addr, const void* gaddr) {
    asm volatile("global_load_async_to_lds_b128 %0, %1, off"
                 :: "v"(lds_addr), "v"(gaddr) : "memory");
}

// ---------------------------------------------------------------------------
// Kernel 0: one-shot fp32 -> bf16 conversion of all GEMM operands, plus a
// transposed copy of memory (mT[b,d,n]) so the output GEMM's B operand has
// contiguous K. Everything stays L2-resident (~13 MB total workspace).
// ---------------------------------------------------------------------------
__global__ void __launch_bounds__(256)
cvt_kernel(const float* __restrict__ q,  const float* __restrict__ m,
           const float* __restrict__ wq, const float* __restrict__ wm,
           unsigned short* __restrict__ qbf,  unsigned short* __restrict__ mbf,
           unsigned short* __restrict__ wqbf, unsigned short* __restrict__ wmbf,
           unsigned short* __restrict__ mTbf)
{
    const int tid    = blockIdx.x * blockDim.x + threadIdx.x;
    const int stride = gridDim.x * blockDim.x;
    for (int i = tid; i < B_ * M_ * D_; i += stride) qbf[i]  = f2bfu(q[i]);
    for (int i = tid; i < H_ * D_;      i += stride) wqbf[i] = f2bfu(wq[i]);
    for (int i = tid; i < H_ * D_;      i += stride) wmbf[i] = f2bfu(wm[i]);
    for (int i = tid; i < B_ * N_ * D_; i += stride) {
        unsigned short v = f2bfu(m[i]);
        mbf[i] = v;
        const int b = i / (N_ * D_);
        const int r = i % (N_ * D_);
        const int n = r / D_, d = r % D_;
        mTbf[((size_t)b * D_ + d) * N_ + n] = v;          // transposed copy
    }
}

// ---------------------------------------------------------------------------
// Kernel 1: fused projection GEMM (bf16 WMMA, f32 accum).
// Rows [0, B*M)       : Qp[row,h] = sum_d query[row,d]  * Wq[h,d]
// Rows [B*M, B*M+B*N) : Mp[row,h] = sum_d memory[row,d] * Wm[h,d]
// Each wave owns a 16x64 output strip: one A fragment reused by 4 B tiles.
// All B fragments preloaded before the WMMA chain to batch loads per clause.
// ---------------------------------------------------------------------------
__global__ void __launch_bounds__(256)
proj_kernel(const unsigned short* __restrict__ qbf,
            const unsigned short* __restrict__ mbf,
            const unsigned short* __restrict__ wqbf,
            const unsigned short* __restrict__ wmbf,
            float* __restrict__ Qp, float* __restrict__ Mp)
{
    const int lane = threadIdx.x & 31;
    const int wave = threadIdx.x >> 5;
    const int tile = blockIdx.x * 8 + wave;          // 1536 waves total
    const int rowTile  = tile >> 3;                  // 0..191
    const int colGroup = tile & 7;                   // 8 groups of 64 columns

    const __bf16* src; const __bf16* W; float* dst; int rowBase;
    if (rowTile < (B_ * M_) / 16) {                  // query rows
        src = (const __bf16*)qbf; W = (const __bf16*)wqbf;
        dst = Qp; rowBase = rowTile * 16;
    } else {                                         // memory rows
        src = (const __bf16*)mbf; W = (const __bf16*)wmbf;
        dst = Mp; rowBase = (rowTile - (B_ * M_) / 16) * 16;
    }

    // A layout (matches B): lanes<16 -> K 0..15; lanes>=16 -> K 16..31
    const int r  = rowBase + (lane & 15);
    const int ka = (lane >> 4) * 16;                 // A chunk base
    const int kb = (lane >> 4) * 16;                 // B half: K 0..15 / 16..31
    const int h0 = colGroup * 64 + (lane & 15);

    v8f c[4] = {};
#pragma unroll 2
    for (int k0 = 0; k0 < D_; k0 += 32) {
        const __bf16* ap = src + (size_t)r * D_ + k0 + ka;
        v8bf lo = *(const v8bf*)ap;
        v8bf hi = *(const v8bf*)(ap + 8);
        v16bf a = __builtin_shufflevector(lo, hi,
                      0,1,2,3,4,5,6,7,8,9,10,11,12,13,14,15);
        v16bf bmv[4];
#pragma unroll
        for (int t = 0; t < 4; ++t)
            bmv[t] = *(const v16bf*)(W + (size_t)(h0 + t * 16) * D_ + k0 + kb);
#pragma unroll
        for (int t = 0; t < 4; ++t)
            c[t] = __builtin_amdgcn_wmma_f32_16x16x32_bf16(
                       false, a, false, bmv[t], (short)0, c[t], false, false);
    }

    // C layout: VGPR i -> row = rowBase + i + (lane>=16 ? 8 : 0)
    const int cr = rowBase + (lane >> 4) * 8;
#pragma unroll
    for (int t = 0; t < 4; ++t)
#pragma unroll
        for (int i = 0; i < 8; ++i)
            dst[(size_t)(cr + i) * H_ + h0 + t * 16] = c[t][i];
}

// ---------------------------------------------------------------------------
// Kernel 2 (dominant cost): per (b,m) row -> aligns over n, softmax, bf16
// scores. The m-projection rows stream through the CDNA5 async global->LDS
// path with per-wave double buffering: next row's 2 KB is in flight while the
// current row's tanh-dot executes on the VALU/TRANS pipes (zero VGPR cost).
// ---------------------------------------------------------------------------
__global__ void __launch_bounds__(256)
align_softmax_kernel(const float* __restrict__ Qp, const float* __restrict__ Mp,
                     const float* __restrict__ w_out,
                     unsigned short* __restrict__ Sb)
{
    const int bm = blockIdx.x;          // [0, B*M)
    const int b  = bm / M_;
    __shared__ __align__(16) float qs[H_];
    __shared__ __align__(16) float ws[H_];
    __shared__ __align__(16) float mbuf[8][2][H_];   // 32 KB double buffers
    __shared__ float al[N_];
    __shared__ float red[8];

    const int tid  = threadIdx.x;       // 256 threads = 8 waves
    const int lane = tid & 31;
    const int wave = tid >> 5;

    const float4* qg = (const float4*)(Qp + (size_t)bm * H_);
    const float4* wg = (const float4*)w_out;
    for (int i = tid; i < H_ / 4; i += 256) {
        ((float4*)qs)[i] = qg[i];
        ((float4*)ws)[i] = wg[i];
    }

    const float* Mb = Mp + (size_t)b * N_ * H_;
    const unsigned lds0 = (unsigned)(uintptr_t)&mbuf[wave][0][0] + lane * 16;
    const unsigned lds1 = (unsigned)(uintptr_t)&mbuf[wave][1][0] + lane * 16;

    // kick off first row for this wave (4 x b128 = 2 KB)
    {
        const char* g = (const char*)(Mb + (size_t)wave * H_) + lane * 16;
#pragma unroll
        for (int j = 0; j < 4; ++j)
            async_copy_b128(lds0 + j * 512, g + j * 512);
    }
    __syncthreads();                    // qs/ws visible to all waves

    const float4* q4 = (const float4*)qs;
    const float4* w4 = (const float4*)ws;
    int cur = 0;
    for (int n = wave; n < N_; n += 8) {
        const int nn = n + 8;
        if (nn < N_) {                  // prefetch next row into other buffer
            const unsigned nxt = cur ? lds0 : lds1;
            const char* g = (const char*)(Mb + (size_t)nn * H_) + lane * 16;
#pragma unroll
            for (int j = 0; j < 4; ++j)
                async_copy_b128(nxt + j * 512, g + j * 512);
            // current buffer's 4 copies done; next 4 may remain in flight
            asm volatile("s_wait_asynccnt 0x4" ::: "memory");
        } else {
            asm volatile("s_wait_asynccnt 0x0" ::: "memory");
        }

        const float4* m4 = (const float4*)&mbuf[wave][cur][0];
        float acc = 0.f;
#pragma unroll
        for (int j = 0; j < H_ / 128; ++j) {
            const int idx = lane + 32 * j;
            const float4 mv = m4[idx];
            const float4 qv = q4[idx];
            const float4 wv = w4[idx];
            acc = fmaf(tanh_fast(qv.x + mv.x), wv.x, acc);
            acc = fmaf(tanh_fast(qv.y + mv.y), wv.y, acc);
            acc = fmaf(tanh_fast(qv.z + mv.z), wv.z, acc);
            acc = fmaf(tanh_fast(qv.w + mv.w), wv.w, acc);
        }
#pragma unroll
        for (int off = 16; off; off >>= 1) acc += __shfl_xor(acc, off, 32);
        if (lane == 0) al[n] = acc;
        cur ^= 1;
    }
    __syncthreads();

    // Softmax over N_=512 (2 elements per thread)
    const float v0 = al[tid], v1 = al[tid + 256];
    float m = fmaxf(v0, v1);
#pragma unroll
    for (int off = 16; off; off >>= 1) m = fmaxf(m, __shfl_xor(m, off, 32));
    if (lane == 0) red[wave] = m;
    __syncthreads();
    float gm = red[0];
#pragma unroll
    for (int w = 1; w < 8; ++w) gm = fmaxf(gm, red[w]);

    const float e0 = __builtin_amdgcn_exp2f((v0 - gm) * 1.442695041f);
    const float e1 = __builtin_amdgcn_exp2f((v1 - gm) * 1.442695041f);
    float s = e0 + e1;
#pragma unroll
    for (int off = 16; off; off >>= 1) s += __shfl_xor(s, off, 32);
    __syncthreads();                    // done reading red[] for gm
    if (lane == 0) red[wave] = s;
    __syncthreads();
    float gs = 0.f;
#pragma unroll
    for (int w = 0; w < 8; ++w) gs += red[w];
    const float inv = __builtin_amdgcn_rcpf(gs);

    Sb[(size_t)bm * N_ + tid]       = f2bfu(e0 * inv);
    Sb[(size_t)bm * N_ + tid + 256] = f2bfu(e1 * inv);
}

// ---------------------------------------------------------------------------
// Kernel 3: out[b,m,d] = sum_n scores[b,m,n] * memory[b,n,d]  (bf16 WMMA).
// A = scores (K=n contiguous); B = transposed memory mT[b,d,n] so each lane's
// column fragment is a contiguous 32B load. 16x64 strip per wave.
// ---------------------------------------------------------------------------
__global__ void __launch_bounds__(256)
out_gemm_kernel(const unsigned short* __restrict__ Sb,
                const unsigned short* __restrict__ mTbf,
                float* __restrict__ out)
{
    const int lane = threadIdx.x & 31;
    const int wave = threadIdx.x >> 5;
    const int tile = blockIdx.x * 8 + wave;          // 512 waves total
    const int dGroups = D_ / 64;                     // 8
    const int mTiles  = M_ / 16;                     // 16
    const int b   = tile / (dGroups * mTiles);
    const int rem = tile % (dGroups * mTiles);
    const int mT  = rem / dGroups;
    const int dG  = rem % dGroups;

    const __bf16* S  = (const __bf16*)Sb + (size_t)b * M_ * N_;
    const __bf16* MT = (const __bf16*)mTbf + (size_t)b * D_ * N_;

    const int r  = mT * 16 + (lane & 15);
    const int d0 = dG * 64 + (lane & 15);
    const int ka = (lane >> 4) * 16;
    const int kb = (lane >> 4) * 16;

    v8f c[4] = {};
#pragma unroll 2
    for (int k0 = 0; k0 < N_; k0 += 32) {
        const __bf16* ap = S + (size_t)r * N_ + k0 + ka;
        v8bf lo = *(const v8bf*)ap;
        v8bf hi = *(const v8bf*)(ap + 8);
        v16bf a = __builtin_shufflevector(lo, hi,
                      0,1,2,3,4,5,6,7,8,9,10,11,12,13,14,15);
        v16bf bmv[4];
#pragma unroll
        for (int t = 0; t < 4; ++t)
            bmv[t] = *(const v16bf*)(MT + (size_t)(d0 + t * 16) * N_ + k0 + kb);
#pragma unroll
        for (int t = 0; t < 4; ++t)
            c[t] = __builtin_amdgcn_wmma_f32_16x16x32_bf16(
                       false, a, false, bmv[t], (short)0, c[t], false, false);
    }

    const int cr = mT * 16 + (lane >> 4) * 8;
#pragma unroll
    for (int t = 0; t < 4; ++t)
#pragma unroll
        for (int i = 0; i < 8; ++i)
            out[((size_t)b * M_ + cr + i) * D_ + d0 + t * 16] = c[t][i];
}

// ---------------------------------------------------------------------------
extern "C" void kernel_launch(void* const* d_in, const int* in_sizes, int n_in,
                              void* d_out, int out_size, void* d_ws, size_t ws_size,
                              hipStream_t stream)
{
    const float* query  = (const float*)d_in[0];
    const float* memory = (const float*)d_in[1];
    const float* Wq     = (const float*)d_in[2];
    const float* Wm     = (const float*)d_in[3];
    const float* w_out  = (const float*)d_in[4];
    float* out = (float*)d_out;

    char* ws = (char*)d_ws;
    size_t off = 0;
    float* Qp = (float*)(ws + off);          off += (size_t)B_ * M_ * H_ * 4;  // 2 MB
    float* Mp = (float*)(ws + off);          off += (size_t)B_ * N_ * H_ * 4;  // 4 MB
    unsigned short* Sbb  = (unsigned short*)(ws + off); off += (size_t)B_ * M_ * N_ * 2; // 1 MB
    unsigned short* qbf  = (unsigned short*)(ws + off); off += (size_t)B_ * M_ * D_ * 2; // 1 MB
    unsigned short* mbf  = (unsigned short*)(ws + off); off += (size_t)B_ * N_ * D_ * 2; // 2 MB
    unsigned short* wqbf = (unsigned short*)(ws + off); off += (size_t)H_ * D_ * 2;      // .5 MB
    unsigned short* wmbf = (unsigned short*)(ws + off); off += (size_t)H_ * D_ * 2;      // .5 MB
    unsigned short* mTbf = (unsigned short*)(ws + off); off += (size_t)B_ * N_ * D_ * 2; // 2 MB

    cvt_kernel<<<2048, 256, 0, stream>>>(query, memory, Wq, Wm,
                                         qbf, mbf, wqbf, wmbf, mTbf);
    proj_kernel<<<192, 256, 0, stream>>>(qbf, mbf, wqbf, wmbf, Qp, Mp);
    align_softmax_kernel<<<B_ * M_, 256, 0, stream>>>(Qp, Mp, w_out, Sbb);
    out_gemm_kernel<<<64, 256, 0, stream>>>(Sbb, mTbf, out);
}
